// DeformableDecoderLayer_28913719837328
// MI455X (gfx1250) — compile-verified, matrix operand order checked by reading, not run on hardware
//
#include <hip/hip_runtime.h>
#include <hip/hip_bf16.h>
#include <math.h>

typedef __attribute__((ext_vector_type(16))) _Float16 v16h;
typedef __attribute__((ext_vector_type(8)))  float    v8f;
typedef unsigned int uv4 __attribute__((ext_vector_type(4)));
typedef int          iv4 __attribute__((ext_vector_type(4)));
typedef int          iv8 __attribute__((ext_vector_type(8)));

#define Bc   8
#define Qc   300
#define Dc   256
#define Hc   8
#define HDc  32
#define Fc   1024
#define NFc  21760

// ---------------------------------------------------------------------------
// f32 -> f16 weight pre-conversion (run once per weight matrix).
// ---------------------------------------------------------------------------
__global__ __launch_bounds__(256)
void k_cvt_f16(const float* __restrict__ src, _Float16* __restrict__ dst, long n) {
  long i = (long)blockIdx.x * 256 + threadIdx.x;
  if (i < n) dst[i] = (_Float16)src[i];
}

// ---------------------------------------------------------------------------
// TDM: DMA one [tile_dim1=128 rows] x [tile_dim0=32 halves] tile of a row-major
// (Nrows x K) f16 tensor into LDS. Descriptor per CDNA5 ISA ch.8:
//   group0: count=1 | lds_addr | global_addr(57b) | type=2
//   group1: data_size=1(2B), tensor_dim0=K, tensor_dim1=Nrows,
//           tile_dim0=32, tile_dim1=128, tensor_dim0_stride=K
// Tracked on TENSORcnt; EXEC ignored; issued by one wave per block.
// This toolchain exposes the 6-arg builtin (g0, g1, g2, g3, g4, cpol).
// ---------------------------------------------------------------------------
__device__ __forceinline__
void tdm_load_b_tile(const _Float16* gsrc, unsigned ldsOff, int K, int Nrows) {
  unsigned long long ga = (unsigned long long)(uintptr_t)gsrc;
  uv4 g0;
  g0[0] = 1u;                                   // count=1 (valid user D#)
  g0[1] = ldsOff;                               // LDS byte address
  g0[2] = (unsigned)ga;                         // global_addr[31:0]
  g0[3] = (unsigned)((ga >> 32) & 0x1FFFFFFu) | 0x80000000u; // addr[56:32] | type=2
  unsigned long long q0 = (1ull << 16)                                   // data_size=2B
                        | ((unsigned long long)((unsigned)K & 0xFFFFu) << 48); // dim0[15:0]
  unsigned long long q1 = ((unsigned long long)((unsigned)K >> 16))      // dim0[31:16]
                        | ((unsigned long long)(unsigned)Nrows << 16)    // tensor_dim1
                        | (32ull << 48);                                 // tile_dim0=32
  unsigned long long q2 = 128ull                                         // tile_dim1=128
                        | ((unsigned long long)(unsigned)K << 32);       // dim0_stride lo
  unsigned long long q3 = 0ull;                                          // stride hi, dim1_stride=0
  iv8 g1;
  g1[0] = (int)(unsigned)q0; g1[1] = (int)(q0 >> 32);
  g1[2] = (int)(unsigned)q1; g1[3] = (int)(q1 >> 32);
  g1[4] = (int)(unsigned)q2; g1[5] = (int)(q2 >> 32);
  g1[6] = (int)(unsigned)q3; g1[7] = (int)(q3 >> 32);
  iv4 z4 = {0, 0, 0, 0};
  iv8 z8 = {0, 0, 0, 0, 0, 0, 0, 0};
  __builtin_amdgcn_tensor_load_to_lds(g0, g1, z4, z4, z8, 0);
}

// ---------------------------------------------------------------------------
// WMMA GEMM: C[M,N] = act(A[M,K]f32 @ W16[N,K]f16^T + bias[N]) (+ residual)
// Block = 8 waves = 128x128 tile; wave owns 16 rows x 128 cols (8 v8f accs).
// B tile (128x32 f16, 8KB) staged in LDS by the Tensor Data Mover, double
// buffered: wait(tensorcnt)->barrier->issue(next)->compute->barrier.
// All 16 ds_load_b128 B-fragment loads are issued into distinct registers
// BEFORE the 8 WMMAs so LDS latency overlaps WMMA issue (no per-WMMA
// s_wait_dscnt 0). A is loaded f32 from global, converted in-register.
// M%16==0, N%128==0, K%32==0 for all GEMMs in this layer. Tail M-blocks use a
// wave-uniform 'active' flag (EXEC all-ones for every wave reaching WMMA).
// ---------------------------------------------------------------------------
__global__ __launch_bounds__(256)
void k_gemm_wmma(const float* __restrict__ A, const _Float16* __restrict__ W16,
                 const float* __restrict__ bias, const float* __restrict__ res,
                 float* __restrict__ C, int M, int N, int K, int relu) {
  __shared__ _Float16 sB[2][128 * 32];          // 2 x 8KB double buffer
  const int wave = threadIdx.x >> 5;
  const int lane = threadIdx.x & 31;
  const int tm = blockIdx.x * 128 + wave * 16;  // wave's 16-row slab
  const int tn = blockIdx.y * 128;              // block's 128-col slab
  const bool active = tm < M;                   // wave-uniform
  const int row = lane & 15;
  const int kb  = (lane >> 4) << 3;             // 0 or 8

  v8f zero = {0.f,0.f,0.f,0.f,0.f,0.f,0.f,0.f};
  v8f acc[8];
#pragma unroll
  for (int j = 0; j < 8; ++j) acc[j] = zero;

  const float* arow = A + (size_t)((active ? tm : 0) + row) * K;
  const _Float16* wtile = W16 + (size_t)tn * K;
  const unsigned ldsOff[2] = { (unsigned)(uintptr_t)&sB[0][0],
                               (unsigned)(uintptr_t)&sB[1][0] };
  const int nk = K >> 5;

  if (wave == 0) tdm_load_b_tile(wtile, ldsOff[0], K, N);   // prime buffer 0

  for (int i = 0; i < nk; ++i) {
    const int k0 = i << 5;
    if (wave == 0) __builtin_amdgcn_s_wait_tensorcnt(0);    // buf[i&1] landed
    __syncthreads();                                        // publish to block
    if (wave == 0 && i + 1 < nk)                            // prefetch buf[(i+1)&1]
      tdm_load_b_tile(wtile + ((size_t)(i + 1) << 5), ldsOff[(i + 1) & 1], K, N);

    if (active) {
      __builtin_prefetch(arow + k0 + 32, 0, 0);             // next A chunk
      v16h a;
#pragma unroll
      for (int e = 0; e < 8; ++e) {
        a[e]     = (_Float16)arow[k0 + kb + e];
        a[e + 8] = (_Float16)arow[k0 + kb + 16 + e];
      }
      const _Float16* bb = &sB[i & 1][0];
      v16h bf[8];                                           // distinct regs per subtile
#pragma unroll
      for (int j = 0; j < 8; ++j) {
        const _Float16* br = bb + (j * 16 + row) * 32;      // ds_load_b128 x2
#pragma unroll
        for (int e = 0; e < 8; ++e) {
          bf[j][e]     = br[kb + e];
          bf[j][e + 8] = br[kb + 16 + e];
        }
      }
#pragma unroll
      for (int j = 0; j < 8; ++j)
        acc[j] = __builtin_amdgcn_wmma_f32_16x16x32_f16(false, a, false, bf[j],
                                                        (short)0, acc[j], false, false);
    }
    __syncthreads();                                        // reads done before TDM reuse
  }

  if (!active) return;
  // C/D layout: col = lane&15, rowM = 8*(lane>=16)+r
  const int cn    = lane & 15;
  const int rbase = (lane >> 4) << 3;
#pragma unroll
  for (int j = 0; j < 8; ++j) {
    const int col = tn + (j << 4) + cn;
    const float bv = bias ? bias[col] : 0.f;
#pragma unroll
    for (int r = 0; r < 8; ++r) {
      const size_t idx = (size_t)(tm + rbase + r) * N + col;
      float v = acc[j][r] + bv;
      if (relu) v = fmaxf(v, 0.f);
      if (res)  v += res[idx];
      C[idx] = v;
    }
  }
}

// ---------------------------------------------------------------------------
// LayerNorm over D=256; one block per row; optional (+pos) second output.
// ---------------------------------------------------------------------------
__global__ __launch_bounds__(256)
void k_layernorm(const float* __restrict__ x, const float* __restrict__ g,
                 const float* __restrict__ b, const float* __restrict__ pos,
                 float* __restrict__ out_ln, float* __restrict__ out_lnp) {
  __shared__ float red[256];
  const size_t base = (size_t)blockIdx.x * Dc;
  const int tid = threadIdx.x;
  const float v = x[base + tid];
  red[tid] = v; __syncthreads();
  for (int s = 128; s > 0; s >>= 1) { if (tid < s) red[tid] += red[tid + s]; __syncthreads(); }
  const float mean = red[0] * (1.0f / Dc); __syncthreads();
  const float c = v - mean;
  red[tid] = c * c; __syncthreads();
  for (int s = 128; s > 0; s >>= 1) { if (tid < s) red[tid] += red[tid + s]; __syncthreads(); }
  const float var = red[0] * (1.0f / Dc);
  const float y = c * rsqrtf(var + 1e-5f) * g[tid] + b[tid];
  if (out_ln)  out_ln[base + tid] = y;
  if (out_lnp) out_lnp[base + tid] = y + (pos ? pos[base + tid] : 0.f);
}

// ---------------------------------------------------------------------------
// Attention scores: att[b,h,q,k] = dot(qh,kh)/sqrt(HD); qkbuf = (B*Q, 512).
// ---------------------------------------------------------------------------
__global__ __launch_bounds__(256)
void k_scores(const float* __restrict__ qkbuf, float* __restrict__ att) {
  const long total = (long)Bc * Hc * Qc * Qc;
  long idx = (long)blockIdx.x * blockDim.x + threadIdx.x;
  if (idx >= total) return;
  int kk = (int)(idx % Qc); long t = idx / Qc;
  int q  = (int)(t % Qc);   t /= Qc;
  int h  = (int)(t % Hc);
  int b  = (int)(t / Hc);
  const float* qr = qkbuf + (size_t)(b * Qc + q)  * 512 + h * HDc;
  const float* kr = qkbuf + (size_t)(b * Qc + kk) * 512 + 256 + h * HDc;
  float acc = 0.f;
#pragma unroll
  for (int d = 0; d < HDc; ++d) acc += qr[d] * kr[d];
  att[idx] = acc * 0.17677669529663687f;   // 1/sqrt(32)
}

// Row softmax, arbitrary row length R (block per row).
__global__ __launch_bounds__(256)
void k_softmax_row(float* __restrict__ x, int R) {
  __shared__ float red[256];
  float* row = x + (size_t)blockIdx.x * R;
  const int tid = threadIdx.x;
  float m = -INFINITY;
  for (int i = tid; i < R; i += 256) m = fmaxf(m, row[i]);
  red[tid] = m; __syncthreads();
  for (int s = 128; s > 0; s >>= 1) { if (tid < s) red[tid] = fmaxf(red[tid], red[tid + s]); __syncthreads(); }
  m = red[0]; __syncthreads();
  float sum = 0.f;
  for (int i = tid; i < R; i += 256) { float e = __expf(row[i] - m); row[i] = e; sum += e; }
  red[tid] = sum; __syncthreads();
  for (int s = 128; s > 0; s >>= 1) { if (tid < s) red[tid] += red[tid + s]; __syncthreads(); }
  const float inv = 1.f / red[0];
  for (int i = tid; i < R; i += 256) row[i] *= inv;
}

// AV: o[b,q,h,d] = sum_k att[b,h,q,k] * v[b,k,h,d]; att slab staged in LDS.
__global__ __launch_bounds__(256)
void k_av(const float* __restrict__ att, const float* __restrict__ vbuf,
          float* __restrict__ out) {
  __shared__ float s_att[Hc * Qc];                 // 2400 floats
  const int bq = blockIdx.x, b = bq / Qc, q = bq % Qc;
  const int tid = threadIdx.x;
  for (int i = tid; i < Hc * Qc; i += 256) {
    int h = i / Qc, k = i % Qc;
    s_att[i] = att[(((size_t)(b * Hc + h) * Qc) + q) * Qc + k];
  }
  __syncthreads();
  const int h = tid >> 5, d = tid & 31;
  float acc = 0.f;
  for (int k = 0; k < Qc; ++k)
    acc += s_att[h * Qc + k] * vbuf[(size_t)(b * Qc + k) * Dc + h * HDc + d];
  out[(size_t)bq * Dc + tid] = acc;
}

// Softmax over L*P=16 per (b,q,h), in place. aw layout (B*Q, H*16).
__global__ __launch_bounds__(256)
void k_softmax16(float* __restrict__ aw) {
  const int i = blockIdx.x * blockDim.x + threadIdx.x;
  if (i >= Bc * Qc * Hc) return;
  float* r = aw + (size_t)i * 16;
  float m = -INFINITY;
#pragma unroll
  for (int j = 0; j < 16; ++j) m = fmaxf(m, r[j]);
  float s = 0.f;
#pragma unroll
  for (int j = 0; j < 16; ++j) { float e = __expf(r[j] - m); r[j] = e; s += e; }
  const float inv = 1.f / s;
#pragma unroll
  for (int j = 0; j < 16; ++j) r[j] *= inv;
}

// Deformable bilinear sampling. Block per (b,q); threads = (h,d).
__global__ __launch_bounds__(256)
void k_deform(const float* __restrict__ ref, const float* __restrict__ off,
              const float* __restrict__ aw, const float* __restrict__ value,
              float* __restrict__ out) {
  __shared__ float s_px[Hc][16], s_py[Hc][16], s_aw[Hc][16];
  const int bq = blockIdx.x, b = bq / Qc;
  const int tid = threadIdx.x;
  if (tid < 128) {
    const int h = tid >> 4, lp = tid & 15, l = lp >> 2, p = lp & 3;
    const int dims[4] = {128, 64, 32, 16};
    const float rx = ref[bq * 2 + 0], ry = ref[bq * 2 + 1];
    const float ox = off[(size_t)bq * Dc + h * 32 + l * 8 + p * 2 + 0];
    const float oy = off[(size_t)bq * Dc + h * 32 + l * 8 + p * 2 + 1];
    s_px[h][lp] = rx * (float)dims[l] + ox - 0.5f;
    s_py[h][lp] = ry * (float)dims[l] + oy - 0.5f;
    s_aw[h][lp] = aw[(size_t)bq * 128 + h * 16 + lp];
  }
  __syncthreads();
  const int h = tid >> 5, d = tid & 31;
  const int dims[4]   = {128, 64, 32, 16};
  const int starts[4] = {0, 16384, 20480, 21504};
  float acc = 0.f;
#pragma unroll
  for (int l = 0; l < 4; ++l) {
    const int Wl = dims[l], Hl = dims[l];
    const int base = b * NFc + starts[l];
#pragma unroll
    for (int p = 0; p < 4; ++p) {
      const float px = s_px[h][l * 4 + p], py = s_py[h][l * 4 + p];
      const float x0 = floorf(px), y0 = floorf(py);
      float s = 0.f;
#pragma unroll
      for (int dy = 0; dy <= 1; ++dy) {
#pragma unroll
        for (int dx = 0; dx <= 1; ++dx) {
          const float xi = x0 + dx, yi = y0 + dy;
          const float w = (1.f - fabsf(px - xi)) * (1.f - fabsf(py - yi));
          if (xi >= 0.f && xi < (float)Wl && yi >= 0.f && yi < (float)Hl && w != 0.f) {
            const int xc = (int)xi, yc = (int)yi;
            s += w * value[(size_t)(base + yc * Wl + xc) * Dc + h * HDc + d];
          }
        }
      }
      acc += s_aw[h][l * 4 + p] * s;
    }
  }
  out[(size_t)bq * Dc + tid] = acc;
}

// ---------------------------------------------------------------------------
extern "C" void kernel_launch(void* const* d_in, const int* in_sizes, int n_in,
                              void* d_out, int out_size, void* d_ws, size_t ws_size,
                              hipStream_t stream) {
  const float* embed  = (const float*)d_in[0];
  const float* pos    = (const float*)d_in[1];
  const float* refp   = (const float*)d_in[2];
  const float* feat   = (const float*)d_in[3];
  const float* ln1_w  = (const float*)d_in[4];
  const float* ln1_b  = (const float*)d_in[5];
  const float* ln2_w  = (const float*)d_in[6];
  const float* ln2_b  = (const float*)d_in[7];
  const float* ln3_w  = (const float*)d_in[8];
  const float* ln3_b  = (const float*)d_in[9];
  const float* inp_w  = (const float*)d_in[10];
  const float* inp_b  = (const float*)d_in[11];
  const float* outp_w = (const float*)d_in[12];   // MHA out proj
  const float* outp_b = (const float*)d_in[13];
  const float* off_w  = (const float*)d_in[14];
  const float* off_b  = (const float*)d_in[15];
  const float* aw_w   = (const float*)d_in[16];
  const float* aw_b   = (const float*)d_in[17];
  const float* val_w  = (const float*)d_in[18];
  const float* val_b  = (const float*)d_in[19];
  const float* dop_w  = (const float*)d_in[20];   // deform out proj
  const float* dop_b  = (const float*)d_in[21];
  const float* f1_w   = (const float*)d_in[22];
  const float* f1_b   = (const float*)d_in[23];
  const float* f2_w   = (const float*)d_in[24];
  const float* f2_b   = (const float*)d_in[25];
  float* out = (float*)d_out;
  float* ws  = (float*)d_ws;

  const int BQ = Bc * Qc;                   // 2400 rows
  const size_t ROW = (size_t)BQ * Dc;       // 614400

  // Workspace layout (floats). Persistent | f16 weights | stage-aliased scratch.
  const size_t EMBED1 = 0;
  const size_t EMBED2 = ROW;
  const size_t W16F   = 2 * ROW;            // 1,015,808 halves = 507,904 floats
  const size_t S      = 2 * ROW + 524288;
  // Stage A (MHA)
  const size_t VLN  = S;
  const size_t QK   = S + ROW;
  const size_t QKB  = S + 2 * ROW;                       // 2400x512
  const size_t VBUF = QKB + (size_t)BQ * 512;
  const size_t ATT  = VBUF + ROW;                        // 8*8*300*300
  const size_t MHAO = ATT + (size_t)Bc * Hc * Qc * Qc;
  // Stage B (deform) — aliases stage A
  const size_t Q2   = S;
  const size_t OFFB = S + ROW;
  const size_t AWB  = S + 2 * ROW;                       // 2400x128
  const size_t DEF  = AWB + (size_t)BQ * 128;
  const size_t VALB = DEF + ROW;                         // 174080x256
  // Stage C (FFN) — aliases stage B
  const size_t X3   = S;
  const size_t FFNH = S + ROW;                           // 2400x1024

  // f16 weight pool offsets (in halves)
  _Float16* w16 = (_Float16*)(ws + W16F);
  const size_t H_INP = 0;        // 768*256
  const size_t H_OUT = 196608;   // 256*256
  const size_t H_OFF = 262144;   // 256*256
  const size_t H_AW  = 327680;   // 128*256
  const size_t H_VAL = 360448;   // 256*256
  const size_t H_DOP = 425984;   // 256*256
  const size_t H_F1  = 491520;   // 1024*256
  const size_t H_F2  = 753664;   // 256*1024

  auto cvt = [&](const float* src, _Float16* dst, long n) {
    hipLaunchKernelGGL(k_cvt_f16, dim3((unsigned)((n + 255) / 256)), dim3(256), 0, stream,
                       src, dst, n);
  };
  auto gemm = [&](const float* A, const _Float16* W, const float* bias, const float* res,
                  float* C, int M, int N, int K, int relu) {
    hipLaunchKernelGGL(k_gemm_wmma, dim3((M + 127) / 128, N / 128), dim3(256), 0, stream,
                       A, W, bias, res, C, M, N, K, relu);
  };

  // ---- Weight pre-conversion (once; ~1M elements total) ----
  cvt(inp_w,  w16 + H_INP, 768 * 256);
  cvt(outp_w, w16 + H_OUT, 256 * 256);
  cvt(off_w,  w16 + H_OFF, 256 * 256);
  cvt(aw_w,   w16 + H_AW,  128 * 256);
  cvt(val_w,  w16 + H_VAL, 256 * 256);
  cvt(dop_w,  w16 + H_DOP, 256 * 256);
  cvt(f1_w,   w16 + H_F1,  1024 * 256);
  cvt(f2_w,   w16 + H_F2,  256 * 1024);

  // ---- Stage A: self-attention ----
  hipLaunchKernelGGL(k_layernorm, dim3(BQ), dim3(256), 0, stream,
                     embed, ln1_w, ln1_b, pos, ws + VLN, ws + QK);
  gemm(ws + QK,  w16 + H_INP,             inp_b,       nullptr, ws + QKB,  BQ, 512, Dc, 0);
  gemm(ws + VLN, w16 + H_INP + 512 * 256, inp_b + 512, nullptr, ws + VBUF, BQ, 256, Dc, 0);
  {
    const long total = (long)Bc * Hc * Qc * Qc;
    hipLaunchKernelGGL(k_scores, dim3((unsigned)((total + 255) / 256)), dim3(256), 0, stream,
                       ws + QKB, ws + ATT);
  }
  hipLaunchKernelGGL(k_softmax_row, dim3(Bc * Hc * Qc), dim3(256), 0, stream, ws + ATT, Qc);
  hipLaunchKernelGGL(k_av, dim3(BQ), dim3(256), 0, stream, ws + ATT, ws + VBUF, ws + MHAO);
  gemm(ws + MHAO, w16 + H_OUT, outp_b, embed, ws + EMBED1, BQ, Dc, Dc, 0);   // + residual

  // ---- Stage B: deformable attention ----
  hipLaunchKernelGGL(k_layernorm, dim3(BQ), dim3(256), 0, stream,
                     ws + EMBED1, ln2_w, ln2_b, pos, (float*)nullptr, ws + Q2);
  gemm(ws + Q2, w16 + H_OFF, off_b, nullptr, ws + OFFB, BQ, Dc,  Dc, 0);
  gemm(ws + Q2, w16 + H_AW,  aw_b,  nullptr, ws + AWB,  BQ, 128, Dc, 0);
  hipLaunchKernelGGL(k_softmax16, dim3((Bc * Qc * Hc + 255) / 256), dim3(256), 0, stream, ws + AWB);
  gemm(feat, w16 + H_VAL, val_b, nullptr, ws + VALB, Bc * NFc, Dc, Dc, 0);   // 22.8 GFLOP
  hipLaunchKernelGGL(k_deform, dim3(BQ), dim3(256), 0, stream,
                     refp, ws + OFFB, ws + AWB, ws + VALB, ws + DEF);
  gemm(ws + DEF, w16 + H_DOP, dop_b, ws + EMBED1, ws + EMBED2, BQ, Dc, Dc, 0); // + residual

  // ---- Stage C: FFN ----
  hipLaunchKernelGGL(k_layernorm, dim3(BQ), dim3(256), 0, stream,
                     ws + EMBED2, ln3_w, ln3_b, (const float*)nullptr, ws + X3, (float*)nullptr);
  gemm(ws + X3,   w16 + H_F1, f1_b, nullptr,     ws + FFNH, BQ, Fc, Dc, 1);    // ReLU
  gemm(ws + FFNH, w16 + H_F2, f2_b, ws + EMBED2, out,       BQ, Dc, Fc, 0);    // + residual
}
